// PSA_Block3D_9964324126840
// MI455X (gfx1250) — matrix operand order repaired and checked by reading, not compiled, not run on hardware
//
#include <hip/hip_runtime.h>
#include <hip/hip_bf16.h>

// MI455X / gfx1250: wave32, WMMA bf16 16x16x32 fp32-acc, TDM async tile loads
// double-buffered against WMMA compute (TENSORcnt pipelining).
typedef __attribute__((ext_vector_type(16))) __bf16 v16bf;
typedef __attribute__((ext_vector_type(8)))  float  v8f;
typedef __attribute__((ext_vector_type(4)))  unsigned int u32x4;
typedef __attribute__((ext_vector_type(8)))  int i32x8;
typedef __attribute__((ext_vector_type(4)))  int i32x4;

#define DIMC 192
#define NHEAD 6
#define HDIM 32
#define HIDC 768
#define LTOK 131072   // 32*64*64
#define NWIN 256      // 4*8*8 windows of 8^3

// ---------- helpers ----------
static __device__ __forceinline__ unsigned short f2bf(float f) {
  unsigned u = __float_as_uint(f);
  return (unsigned short)((u + 0x7FFFu + ((u >> 16) & 1u)) >> 16);  // RNE
}
static __device__ __forceinline__ float bf2f(unsigned short h) {
  return __uint_as_float(((unsigned)h) << 16);
}
static __device__ __forceinline__ float gelu_f(float x) {
  return 0.5f * x * (1.0f + erff(x * 0.70710678118654752f));
}

// 6-arg toolchain form: (g0 u32x4, g1 i32x8, g2 i32x4, g3 i32x4, i32x8, cpol)
static __device__ __forceinline__ void tdm_issue(u32x4 g0, i32x8 g1, i32x4 g2, i32x4 g3) {
  i32x8 z8;
#pragma unroll
  for (int i = 0; i < 8; ++i) z8[i] = 0;
  __builtin_amdgcn_tensor_load_to_lds(g0, g1, g2, g3, z8, 0);
}

// TDM: DMA a 2D tile (tile_h rows x tile_w bf16) global->LDS. ISA ch8 D# layout.
// group0: [1:0]=count=1, [63:32]=lds_addr, [120:64]=global_addr, [127:126]=type 2.
// group1: [17:16]=data_size(1=2B), [79:48]=tensor_dim0, [111:80]=tensor_dim1,
//         [127:112]=tile_dim0, [143:128]=tile_dim1, [207:160]=tensor_dim0_stride.
static __device__ __forceinline__ void tdm_load_2d(unsigned lds_off, const void* gptr,
                                                   unsigned tile_w, unsigned tile_h,
                                                   unsigned tdim0, unsigned tdim1,
                                                   unsigned stride0) {
  unsigned long long ga = (unsigned long long)gptr;
  u32x4 g0;
  g0[0] = 1u;
  g0[1] = lds_off;
  g0[2] = (unsigned)ga;
  g0[3] = (unsigned)((ga >> 32) & 0x1FFFFFFu) | (2u << 30);
  i32x8 g1;
  g1[0] = (int)(1u << 16);
  g1[1] = (int)((tdim0 & 0xFFFFu) << 16);
  g1[2] = (int)(((tdim0 >> 16) & 0xFFFFu) | ((tdim1 & 0xFFFFu) << 16));
  g1[3] = (int)(((tdim1 >> 16) & 0xFFFFu) | (tile_w << 16));
  g1[4] = (int)(tile_h & 0xFFFFu);
  g1[5] = (int)stride0;
  g1[6] = 0; g1[7] = 0;
  i32x4 z; z[0] = z[1] = z[2] = z[3] = 0;
  tdm_issue(g0, g1, z, z);
}

// A-matrix fragment (16xK slice, bf16) from LDS, row-major rows of `stride` u16.
// ISA 7.12.2: lanes 0-15 row=r K={0..7,16..23}; lanes 16-31 row=r K={8..15,24..31}.
static __device__ __forceinline__ v16bf frag_a_lds(const unsigned short* base, int row0,
                                                   int stride, int kbase, int lane) {
  const int hl = lane >> 4, r = lane & 15;
  const unsigned short* rp = base + (size_t)(row0 + r) * stride;
  union { v16bf v; unsigned u[8]; } f;
#pragma unroll
  for (int j = 0; j < 8; ++j) {
    int kl = kbase + 2 * (j & 3) + 16 * (j >> 2) + 8 * hl;
    f.u[j] = *(const unsigned*)(rp + kl);
  }
  return f.v;
}
// B-matrix fragment (Kx16 slice) from LDS stored TRANSPOSED: baseT[col][k].
static __device__ __forceinline__ v16bf frag_b_lds(const unsigned short* baseT, int col0,
                                                   int stride, int kbase, int lane) {
  const int hl = lane >> 4, r = lane & 15;
  const unsigned short* cp = baseT + (size_t)(col0 + r) * stride;
  union { v16bf v; unsigned u[8]; } f;
#pragma unroll
  for (int j = 0; j < 8; ++j) {
    int kl = kbase + 2 * (j & 3) + 16 * (j >> 2) + 8 * hl;
    f.u[j] = *(const unsigned*)(cp + kl);
  }
  return f.v;
}

// ---------- weight conversion (folds q scale hd^-0.5 into W_q, b_q) ----------
__global__ void conv_wts_kernel(const float* qw, const float* qbias, const float* kvw,
                                const float* projw, const float* fc1w, const float* fc2w,
                                unsigned short* qw_bf, float* qb_s, unsigned short* kvw_bf,
                                unsigned short* projw_bf, unsigned short* fc1w_bf,
                                unsigned short* fc2w_bf) {
  const float s = 0.17677669529663687f;  // 32^-0.5
  for (int idx = blockIdx.x * blockDim.x + threadIdx.x; idx < DIMC * HIDC;
       idx += gridDim.x * blockDim.x) {
    fc1w_bf[idx] = f2bf(fc1w[idx]);
    fc2w_bf[idx] = f2bf(fc2w[idx]);
    if (idx < DIMC * DIMC) { qw_bf[idx] = f2bf(qw[idx] * s); projw_bf[idx] = f2bf(projw[idx]); }
    if (idx < DIMC * 48) kvw_bf[idx] = f2bf(kvw[idx]);
    if (idx < DIMC) qb_s[idx] = qbias[idx] * s;
  }
}

// ---------- LayerNorm: one wave per token (wave32), bf16 output ----------
__global__ __launch_bounds__(256) void ln_kernel(const float* __restrict__ x,
                                                 const float* __restrict__ g,
                                                 const float* __restrict__ b,
                                                 unsigned short* __restrict__ out) {
  const int wave = threadIdx.x >> 5, lane = threadIdx.x & 31;
  const size_t t = (size_t)blockIdx.x * 8 + wave;
  const float* xp = x + t * DIMC;
  float v[6]; float s = 0.f;
#pragma unroll
  for (int k = 0; k < 6; ++k) { v[k] = xp[k * 32 + lane]; s += v[k]; }
#pragma unroll
  for (int m = 1; m < 32; m <<= 1) s += __shfl_xor(s, m, 32);
  const float mean = s * (1.0f / DIMC);
  float vs = 0.f;
#pragma unroll
  for (int k = 0; k < 6; ++k) { float d = v[k] - mean; vs += d * d; }
#pragma unroll
  for (int m = 1; m < 32; m <<= 1) vs += __shfl_xor(vs, m, 32);
  const float inv = rsqrtf(vs * (1.0f / DIMC) + 1e-5f);
  unsigned short* op = out + t * DIMC;
#pragma unroll
  for (int k = 0; k < 6; ++k) {
    int c = k * 32 + lane;
    op[c] = f2bf((v[k] - mean) * inv * g[c] + b[c]);
  }
}

// ---------- bf16 GEMM: C = act(A[MxK]@B[KxN] + bias) (+res), out f32/bf16 ----------
// Double-buffered pipeline: TDM DMA of A tile i+1 + B stage i+1 overlap WMMA on
// tile i; one barrier per K-step; TENSORcnt drained before the tile is consumed.
__global__ __launch_bounds__(256) void gemm_bf16_kernel(
    const unsigned short* __restrict__ A, const unsigned short* __restrict__ B,
    const float* __restrict__ bias, const float* __restrict__ residual,
    float* __restrict__ outF, unsigned short* __restrict__ outBf,
    int M, int N, int K, int do_gelu) {
  __shared__ __align__(16) unsigned short aT[2][128][32];  // A tiles (TDM dest)
  __shared__ __align__(16) unsigned short bT[2][64][32];   // B tiles TRANSPOSED [col][k]
  const int tid = threadIdx.x;
  const int wave = tid >> 5, lane = tid & 31;
  const int hl = lane >> 4, r = lane & 15;
  const int wm = wave & 3, wn = wave >> 2;          // 4x2 wave grid -> 128x64 tile
  const int m0 = blockIdx.x * 128, n0 = blockIdx.y * 64;
  const int niter = K >> 5;

  auto stageB = [&](int bi, int k0) {
    int rr = tid >> 3, seg = tid & 7;
    int col0 = n0 + seg * 8;
    union { uint4 v4; unsigned short us[8]; } bv;
    if (col0 < N) bv.v4 = *(const uint4*)(B + (size_t)(k0 + rr) * N + col0);
    else {
#pragma unroll
      for (int j = 0; j < 8; ++j) bv.us[j] = 0;
    }
#pragma unroll
    for (int j = 0; j < 8; ++j) bT[bi][seg * 8 + j][rr] = bv.us[j];
  };

  v8f acc[2][2];
#pragma unroll
  for (int mt = 0; mt < 2; ++mt)
#pragma unroll
    for (int nt = 0; nt < 2; ++nt)
#pragma unroll
      for (int i = 0; i < 8; ++i) acc[mt][nt][i] = 0.0f;

  // prologue: stage tile 0 into buffer 0
  if (tid < 32) {  // one wave issues the DMA (per-wave issue; EXEC ignored by TDM)
    tdm_load_2d((unsigned)(uintptr_t)(&aT[0][0][0]), A + (size_t)m0 * K,
                /*tile_w=*/32, /*tile_h=*/128,
                (unsigned)K, (unsigned)M, (unsigned)K);
  }
  stageB(0, 0);
  if (tid < 32) __builtin_amdgcn_s_wait_tensorcnt(0);
  __syncthreads();

  for (int i = 0; i < niter; ++i) {
    const int cur = i & 1, nxt = cur ^ 1;
    const int k0 = i << 5;
    if (i + 1 < niter) {  // kick off tile i+1 while computing tile i
      if (tid < 32) {
        tdm_load_2d((unsigned)(uintptr_t)(&aT[nxt][0][0]), A + (size_t)m0 * K + k0 + 32,
                    32, 128, (unsigned)K, (unsigned)M, (unsigned)K);
      }
      stageB(nxt, k0 + 32);
      if (k0 + 64 < K)                              // gfx1250 global_prefetch (B stream)
        __builtin_prefetch(B + (size_t)(k0 + 64 + (tid >> 3)) * N + n0, 0, 1);
    }

    v16bf af[2];
#pragma unroll
    for (int mt = 0; mt < 2; ++mt)
      af[mt] = frag_a_lds(&aT[cur][0][0], wm * 32 + mt * 16, 32, 0, lane);
#pragma unroll
    for (int nt = 0; nt < 2; ++nt) {
      v16bf bf = frag_b_lds(&bT[cur][0][0], wn * 32 + nt * 16, 32, 0, lane);
#pragma unroll
      for (int mt = 0; mt < 2; ++mt)
        acc[mt][nt] = __builtin_amdgcn_wmma_f32_16x16x32_bf16(
            false, af[mt], false, bf, (short)0, acc[mt][nt], false, false);
    }

    if (tid < 32 && i + 1 < niter) __builtin_amdgcn_s_wait_tensorcnt(0);
    __syncthreads();
  }

#pragma unroll
  for (int mt = 0; mt < 2; ++mt)
#pragma unroll
    for (int nt = 0; nt < 2; ++nt) {
      int col = n0 + wn * 32 + nt * 16 + r;
      if (col < N) {
#pragma unroll
        for (int i = 0; i < 8; ++i) {
          int row = m0 + wm * 32 + mt * 16 + i + 8 * hl;
          float val = acc[mt][nt][i] + bias[col];
          if (do_gelu) val = gelu_f(val);
          size_t off = (size_t)row * N + col;
          if (residual) val += residual[off];
          if (outF) outF[off] = val;
          if (outBf) outBf[off] = f2bf(val);
        }
      }
    }
}

// ---------- fused window attention: 1 block = (window, head) ----------
__global__ __launch_bounds__(256) void attn_kernel(
    const unsigned short* __restrict__ qb, const unsigned short* __restrict__ kvb,
    const float* __restrict__ btab, unsigned short* __restrict__ ob) {
  __shared__ __align__(16) unsigned short q_lds[512][32];   // 32KB (TDM dest)
  __shared__ __align__(16) unsigned short k_lds[64][32];    // 4KB  (= B^T for q@k^T)
  __shared__ __align__(16) unsigned short vT_lds[32][64];   // 4KB  (= B^T for P@v)
  __shared__ __align__(16) unsigned short p_lds[8][16][64]; // 16KB per-wave P scratch

  const int tid = threadIdx.x;
  const int wave = tid >> 5, lane = tid & 31;
  const int hl = lane >> 4, r = lane & 15;
  const int win = blockIdx.x / NHEAD, head = blockIdx.x % NHEAD;
  const int bd = (win >> 6) * 8, bh = ((win >> 3) & 7) * 8, bw = (win & 7) * 8;

  // q tile (512 x 32) for this head: 4D TDM tile 32ch x 8w x 8h x 8d,
  // strides 192 / 64*192 / 4096*192 elements; DMA overlaps the k/v VALU gather.
  if (tid < 32) {
    size_t base_tok = (size_t)bd * 4096 + (size_t)bh * 64 + bw;
    unsigned long long ga = (unsigned long long)(const void*)(qb + base_tok * DIMC + head * HDIM);
    u32x4 g0;
    g0[0] = 1u;
    g0[1] = (unsigned)(uintptr_t)(&q_lds[0][0]);
    g0[2] = (unsigned)ga;
    g0[3] = (unsigned)((ga >> 32) & 0x1FFFFFFu) | (2u << 30);
    i32x8 g1;
    g1[0] = (int)(1u << 16);                 // data_size = 2B
    g1[1] = (int)(192u << 16);               // tensor_dim0 = 192
    g1[2] = (int)(64u << 16);                // tensor_dim1 = 64
    g1[3] = (int)(32u << 16);                // tile_dim0 = 32 channels
    g1[4] = (int)(8u | (8u << 16));          // tile_dim1 = 8 (w), tile_dim2 = 8 (h)
    g1[5] = (int)192u;                       // tensor_dim0_stride = 192 (next w token)
    g1[6] = (int)(12288u << 16);             // tensor_dim1_stride = 64*192 (next h row)
    g1[7] = 0;
    i32x4 g2;
    g2[0] = 64;                              // tensor_dim2
    g2[1] = 32;                              // tensor_dim3
    g2[2] = 786432;                          // tensor_dim2_stride = 4096*192 (next d)
    g2[3] = (int)(8u << 16);                 // tile_dim3 = 8 (d)
    i32x4 g3; g3[0] = g3[1] = g3[2] = g3[3] = 0;
    tdm_issue(g0, g1, g2, g3);
  }
  // k (64x32) & v^T (32x64): 2x2x2 neighborhood folded into head_dim
#pragma unroll
  for (int qi = 0; qi < 8; ++qi) {
    int it = qi * 256 + tid;
    int m = it >> 5, c = it & 31;
    int dd = (c >> 4) & 1, hh = (c >> 3) & 1, ww = (c >> 2) & 1, e = c & 3;
    int pd = m >> 4, ph = (m >> 2) & 3, pw = m & 3;
    size_t t = (size_t)(bd + pd * 2 + dd) * 4096 + (bh + ph * 2 + hh) * 64 + (bw + pw * 2 + ww);
    size_t base = t * 48 + head * 4 + e;
    k_lds[m][c] = kvb[base];          // kv select 0
    vT_lds[c][m] = kvb[base + 24];    // kv select 1
  }
  if (tid < 32) __builtin_amdgcn_s_wait_tensorcnt(0);
  __syncthreads();

  for (int mt = 0; mt < 4; ++mt) {    // each wave owns 64 query rows
    const int mrow0 = wave * 64 + mt * 16;
    v8f s[4];
#pragma unroll
    for (int nt = 0; nt < 4; ++nt)
#pragma unroll
      for (int i = 0; i < 8; ++i) s[nt][i] = 0.0f;

    v16bf aq = frag_a_lds(&q_lds[0][0], mrow0, 32, 0, lane);   // K = 32 (head dim)
#pragma unroll
    for (int nt = 0; nt < 4; ++nt) {
      v16bf bk = frag_b_lds(&k_lds[0][0], nt * 16, 32, 0, lane);
      s[nt] = __builtin_amdgcn_wmma_f32_16x16x32_bf16(false, aq, false, bk,
                                                      (short)0, s[nt], false, false);
    }

    // bias (from pooled-coord geometry) + softmax over 64 cols (16 lanes x 4 regs)
#pragma unroll
    for (int i = 0; i < 8; ++i) {
      int nq = mrow0 + i + 8 * hl;
      int qd = (nq >> 6) >> 1, qh = ((nq >> 3) & 7) >> 1, qw = (nq & 7) >> 1;
#pragma unroll
      for (int nt = 0; nt < 4; ++nt) {
        int m = nt * 16 + r;
        int kd = m >> 4, kh = (m >> 2) & 3, kw = m & 3;
        int idx = (qd - kd + 3) * 49 + (qh - kh + 3) * 7 + (qw - kw + 3);
        s[nt][i] += btab[idx * NHEAD + head];
      }
      float mx = s[0][i];
#pragma unroll
      for (int nt = 1; nt < 4; ++nt) mx = fmaxf(mx, s[nt][i]);
#pragma unroll
      for (int msk = 1; msk < 16; msk <<= 1) mx = fmaxf(mx, __shfl_xor(mx, msk, 32));
      float sum = 0.0f;
#pragma unroll
      for (int nt = 0; nt < 4; ++nt) { float e = __expf(s[nt][i] - mx); s[nt][i] = e; sum += e; }
#pragma unroll
      for (int msk = 1; msk < 16; msk <<= 1) sum += __shfl_xor(sum, msk, 32);
      float inv = 1.0f / sum;
#pragma unroll
      for (int nt = 0; nt < 4; ++nt)
        p_lds[wave][i + 8 * hl][nt * 16 + r] = f2bf(s[nt][i] * inv);
    }
    __syncthreads();

    v8f o[2];
#pragma unroll
    for (int nt = 0; nt < 2; ++nt)
#pragma unroll
      for (int i = 0; i < 8; ++i) o[nt][i] = 0.0f;
#pragma unroll
    for (int ks = 0; ks < 2; ++ks) {              // K = 64 pooled kv tokens
      v16bf ap = frag_a_lds(&p_lds[wave][0][0], 0, 64, ks * 32, lane);
#pragma unroll
      for (int nt = 0; nt < 2; ++nt) {
        v16bf bv = frag_b_lds(&vT_lds[0][0], nt * 16, 64, ks * 32, lane);
        o[nt] = __builtin_amdgcn_wmma_f32_16x16x32_bf16(false, ap, false, bv,
                                                        (short)0, o[nt], false, false);
      }
    }
#pragma unroll
    for (int i = 0; i < 8; ++i) {                 // scatter O back in token order
      int nq = mrow0 + i + 8 * hl;
      size_t t = (size_t)(bd + (nq >> 6)) * 4096 + (bh + ((nq >> 3) & 7)) * 64 + (bw + (nq & 7));
#pragma unroll
      for (int nt = 0; nt < 2; ++nt)
        ob[t * DIMC + head * HDIM + nt * 16 + r] = f2bf(o[nt][i]);
    }
    __syncthreads();
  }
}

// ---------- depthwise 3x3x3 conv branch: h2 = h + gelu(conv(h)+b) ----------
__global__ __launch_bounds__(256) void dwconv_kernel(const unsigned short* __restrict__ hbuf,
                                                     const float* __restrict__ w,
                                                     const float* __restrict__ b,
                                                     unsigned short* __restrict__ h2) {
  const size_t total = (size_t)LTOK * HIDC;
  for (size_t idx = (size_t)blockIdx.x * blockDim.x + threadIdx.x; idx < total;
       idx += (size_t)gridDim.x * blockDim.x) {
    int c = (int)(idx % HIDC);
    size_t t = idx / HIDC;
    int iw = (int)(t & 63), ih = (int)((t >> 6) & 63), id = (int)(t >> 12);
    float acc = 0.f;
#pragma unroll
    for (int kd = 0; kd < 3; ++kd) {
      int dd = id + kd - 1;
      if ((unsigned)dd >= 32u) continue;
#pragma unroll
      for (int kh = 0; kh < 3; ++kh) {
        int hh = ih + kh - 1;
        if ((unsigned)hh >= 64u) continue;
#pragma unroll
        for (int kw = 0; kw < 3; ++kw) {
          int ww = iw + kw - 1;
          if ((unsigned)ww >= 64u) continue;
          size_t tn = ((size_t)dd * 64 + hh) * 64 + ww;
          acc += bf2f(hbuf[tn * HIDC + c]) * w[((kd * 3 + kh) * 3 + kw) * HIDC + c];
        }
      }
    }
    float hv = bf2f(hbuf[t * HIDC + c]);
    h2[t * HIDC + c] = f2bf(hv + gelu_f(acc + b[c]));
  }
}

extern "C" void kernel_launch(void* const* d_in, const int* in_sizes, int n_in,
                              void* d_out, int out_size, void* d_ws, size_t ws_size,
                              hipStream_t stream) {
  (void)in_sizes; (void)n_in; (void)out_size; (void)ws_size;
  const float* x      = (const float*)d_in[0];
  const float* n1g    = (const float*)d_in[1];
  const float* n1b    = (const float*)d_in[2];
  const float* q_w    = (const float*)d_in[3];
  const float* q_b    = (const float*)d_in[4];
  const float* kv_w   = (const float*)d_in[5];
  const float* kv_b   = (const float*)d_in[6];
  const float* btab   = (const float*)d_in[7];
  const float* proj_w = (const float*)d_in[8];
  const float* proj_b = (const float*)d_in[9];
  const float* n2g    = (const float*)d_in[10];
  const float* n2b    = (const float*)d_in[11];
  const float* fc1_w  = (const float*)d_in[12];
  const float* fc1_b  = (const float*)d_in[13];
  const float* dw_w   = (const float*)d_in[14];
  const float* dw_b   = (const float*)d_in[15];
  const float* fc2_w  = (const float*)d_in[16];
  const float* fc2_b  = (const float*)d_in[17];
  // d_in[18..20] = d,h,w scalars (fixed 32,64,64 per reference constants)

  char* wsp = (char*)d_ws;
  size_t off = 0;
  auto carve = [&](size_t bytes) -> void* {
    void* p = wsp + off;
    off += (bytes + 255) & ~(size_t)255;
    return p;
  };
  unsigned short* xn    = (unsigned short*)carve((size_t)LTOK * DIMC * 2);
  unsigned short* qbuf  = (unsigned short*)carve((size_t)LTOK * DIMC * 2);
  unsigned short* kvbuf = (unsigned short*)carve((size_t)LTOK * 48 * 2);
  unsigned short* obuf  = (unsigned short*)carve((size_t)LTOK * DIMC * 2);
  float*          x1    = (float*)carve((size_t)LTOK * DIMC * 4);
  unsigned short* hbuf  = (unsigned short*)carve((size_t)LTOK * HIDC * 2);
  unsigned short* h2buf = (unsigned short*)carve((size_t)LTOK * HIDC * 2);
  unsigned short* qw_bf    = (unsigned short*)carve((size_t)DIMC * DIMC * 2);
  unsigned short* kvw_bf   = (unsigned short*)carve((size_t)DIMC * 48 * 2);
  unsigned short* projw_bf = (unsigned short*)carve((size_t)DIMC * DIMC * 2);
  unsigned short* fc1w_bf  = (unsigned short*)carve((size_t)DIMC * HIDC * 2);
  unsigned short* fc2w_bf  = (unsigned short*)carve((size_t)HIDC * DIMC * 2);
  float*          qb_s     = (float*)carve(DIMC * 4);

  conv_wts_kernel<<<576, 256, 0, stream>>>(q_w, q_b, kv_w, proj_w, fc1_w, fc2_w,
                                           qw_bf, qb_s, kvw_bf, projw_bf, fc1w_bf, fc2w_bf);
  ln_kernel<<<LTOK / 8, 256, 0, stream>>>(x, n1g, n1b, xn);

  dim3 g192(LTOK / 128, DIMC / 64);
  gemm_bf16_kernel<<<g192, 256, 0, stream>>>(xn, qw_bf, qb_s, nullptr, nullptr, qbuf,
                                             LTOK, DIMC, DIMC, 0);
  dim3 g48(LTOK / 128, 1);
  gemm_bf16_kernel<<<g48, 256, 0, stream>>>(xn, kvw_bf, kv_b, nullptr, nullptr, kvbuf,
                                            LTOK, 48, DIMC, 0);
  attn_kernel<<<NWIN * NHEAD, 256, 0, stream>>>(qbuf, kvbuf, btab, obuf);
  gemm_bf16_kernel<<<g192, 256, 0, stream>>>(obuf, projw_bf, proj_b, x, x1, nullptr,
                                             LTOK, DIMC, DIMC, 0);
  ln_kernel<<<LTOK / 8, 256, 0, stream>>>(x1, n2g, n2b, xn);   // reuse xn for LN2
  dim3 g768(LTOK / 128, HIDC / 64);
  gemm_bf16_kernel<<<g768, 256, 0, stream>>>(xn, fc1w_bf, fc1_b, nullptr, nullptr, hbuf,
                                             LTOK, HIDC, DIMC, 1);
  dwconv_kernel<<<8192, 256, 0, stream>>>(hbuf, dw_w, dw_b, h2buf);
  gemm_bf16_kernel<<<g192, 256, 0, stream>>>(h2buf, fc2w_bf, fc2_b, x1, (float*)d_out, nullptr,
                                             LTOK, DIMC, HIDC, 0);
}
